// Head_32959579029766
// MI455X (gfx1250) — compile-verified, matrix operand order checked by reading, not run on hardware
//
#include <hip/hip_runtime.h>
#include <hip/hip_bf16.h>

// ---------------------------------------------------------------------------
// Causal single-head attention for MI455X (gfx1250, wave32, WMMA).
//   q,k,v = f16(x @ W + b)   (score scale sqrt(64)=8 folded into q)
//   out   = softmax_causal(q k^T) v   (fp32 accumulate throughout)
// K-tile staging uses GLOBAL_LOAD_ASYNC_TO_LDS_B128 (ASYNCcnt path).
// ---------------------------------------------------------------------------

typedef __attribute__((ext_vector_type(16))) _Float16 v16h;
typedef __attribute__((ext_vector_type(8)))  _Float16 v8h;
typedef __attribute__((ext_vector_type(8)))  float    v8f;
typedef __attribute__((ext_vector_type(4)))  float    v4f;

#define HEADSZ   64
#define NEMBD    1024
#define SEQT     2048
#define NB       4
#define NROWS    (NB * SEQT)       // 8192 flat rows
#define LDS_STR  72                // halves: 144B row stride -> 16B-aligned b128,
                                   // 36-dword stride is conflict-free over 16 lanes

static __device__ __forceinline__ v16h pack16(v8h lo, v8h hi) {
    v16h r;
#pragma unroll
    for (int j = 0; j < 8; ++j) { r[j] = lo[j]; r[8 + j] = hi[j]; }
    return r;
}

// ---------------------------------------------------------------------------
// Kernel 1: QKV projection.  grid=128 WGs x 128 thr; wave = 16 rows of x.
// A operand = x tile (global, fp32->f16); B operands = W slice staged in LDS
// transposed to [n][k] f16 so each B build is two aligned ds_load_b128.
// ---------------------------------------------------------------------------
__global__ __launch_bounds__(128) void qkv_proj_wmma(
    const float* __restrict__ x,
    const float* __restrict__ Wq, const float* __restrict__ bq,
    const float* __restrict__ Wk, const float* __restrict__ bk,
    const float* __restrict__ Wv, const float* __restrict__ bv,
    _Float16* __restrict__ qh, _Float16* __restrict__ kh, _Float16* __restrict__ vh)
{
    __shared__ _Float16 ldsW[3][64][LDS_STR];   // [mat][n][kk]

    const int tid  = threadIdx.x;
    const int wave = tid >> 5;
    const int lane = tid & 31;
    const int l15  = lane & 15;
    const int lhi  = lane >> 4;                 // 0 or 1
    const int m0   = blockIdx.x * 64 + wave * 16;

    const float* W[3] = { Wq, Wk, Wv };

    v8f acc[3][4];
#pragma unroll
    for (int m = 0; m < 3; ++m)
#pragma unroll
        for (int t = 0; t < 4; ++t) acc[m][t] = (v8f)0.0f;

    const int kk_t = tid >> 2;                  // 0..31 : K row in slice
    const int nb_t = (tid & 3) * 16;            // 0,16,32,48 : N base

    for (int k0 = 0; k0 < NEMBD; k0 += 32) {
        __syncthreads();                        // previous tile fully consumed
        // ---- stage W[k0..k0+31][0..63] (q,k,v) transposed into LDS as f16 ----
#pragma unroll
        for (int m = 0; m < 3; ++m) {
            const float* src = W[m] + (size_t)(k0 + kk_t) * HEADSZ + nb_t;
            v4f w0 = *(const v4f*)(src + 0);
            v4f w1 = *(const v4f*)(src + 4);
            v4f w2 = *(const v4f*)(src + 8);
            v4f w3 = *(const v4f*)(src + 12);
            float tmp[16];
            *(v4f*)(tmp + 0)  = w0; *(v4f*)(tmp + 4)  = w1;
            *(v4f*)(tmp + 8)  = w2; *(v4f*)(tmp + 12) = w3;
#pragma unroll
            for (int j = 0; j < 16; ++j)
                ldsW[m][nb_t + j][kk_t] = (_Float16)tmp[j];
        }

        // ---- A operand: x rows m0..m0+15, K = k0..k0+31 (WMMA A 16x32 layout) ----
        const float* xp = x + (size_t)(m0 + l15) * NEMBD + k0 + lhi * 8;
        v4f x0 = *(const v4f*)(xp + 0);
        v4f x1 = *(const v4f*)(xp + 4);
        v4f x2 = *(const v4f*)(xp + 16);
        v4f x3 = *(const v4f*)(xp + 20);
        v16h a;
#pragma unroll
        for (int j = 0; j < 4; ++j) {
            a[j]      = (_Float16)x0[j];
            a[4 + j]  = (_Float16)x1[j];
            a[8 + j]  = (_Float16)x2[j];
            a[12 + j] = (_Float16)x3[j];
        }
        __syncthreads();                        // staging visible

        // ---- 12 WMMAs: 3 matrices x 4 N-tiles ----
#pragma unroll
        for (int m = 0; m < 3; ++m) {
#pragma unroll
            for (int t = 0; t < 4; ++t) {
                const _Float16* bp = &ldsW[m][t * 16 + l15][lhi * 8];
                v16h b = pack16(*(const v8h*)bp, *(const v8h*)(bp + 16));
                acc[m][t] = __builtin_amdgcn_wmma_f32_16x16x32_f16(
                    false, a, false, b, (short)0, acc[m][t], false, false);
            }
        }
    }

    // ---- epilogue: bias, fold score scale (sqrt(64)=8) into q, write f16 ----
    const float* bias[3] = { bq, bk, bv };
    _Float16*    dst[3]  = { qh, kh, vh };
#pragma unroll
    for (int m = 0; m < 3; ++m) {
#pragma unroll
        for (int t = 0; t < 4; ++t) {
            const int   n  = t * 16 + l15;
            const float bb = bias[m][n];
            const float sc = (m == 0) ? 8.0f : 1.0f;
#pragma unroll
            for (int r = 0; r < 8; ++r) {
                const int row = m0 + r + 8 * lhi;          // C layout: M=r+8*lhi
                dst[m][(size_t)row * HEADSZ + n] = (_Float16)((acc[m][t][r] + bb) * sc);
            }
        }
    }
}

// ---------------------------------------------------------------------------
// Kernel 2: flash attention.  grid = 4 batches * 32 WGs; wave = 16 q rows,
// WG = 64 q rows.  64-key tiles: K staged row-major straight into LDS with
// async-to-LDS b128 copies (ASYNCcnt), V transposed through VGPRs (needs a
// transform).  Online softmax via shfl_xor row reductions in 16-lane halves.
// ---------------------------------------------------------------------------
__global__ __launch_bounds__(128) void flash_attn_wmma(
    const _Float16* __restrict__ qh, const _Float16* __restrict__ kh,
    const _Float16* __restrict__ vh, float* __restrict__ out)
{
    __shared__ _Float16 ldsK [64][LDS_STR];     // [key][head]
    __shared__ _Float16 ldsVt[64][LDS_STR];     // [head][key]
    __shared__ _Float16 ldsP [4][16][LDS_STR];  // per-wave P transpose patch

    const int tid  = threadIdx.x;
    const int wave = tid >> 5;
    const int lane = tid & 31;
    const int l15  = lane & 15;
    const int lhi  = lane >> 4;

    const int    batch = blockIdx.x >> 5;       // 32 WGs per batch
    const int    wg0   = (blockIdx.x & 31) * 64;
    const int    q0    = wg0 + wave * 16;
    const size_t rbase = (size_t)batch * SEQT;

    // Q A-operands, head 0..31 and 32..63 (scale pre-folded)
    const _Float16* qp = qh + (rbase + q0 + l15) * HEADSZ + lhi * 8;
    v16h qa0 = pack16(*(const v8h*)(qp + 0),  *(const v8h*)(qp + 16));
    v16h qa1 = pack16(*(const v8h*)(qp + 32), *(const v8h*)(qp + 48));

    v8f oacc[4];
#pragma unroll
    for (int t = 0; t < 4; ++t) oacc[t] = (v8f)0.0f;
    float mrow[8], lsum[8];
#pragma unroll
    for (int r = 0; r < 8; ++r) { mrow[r] = -1e30f; lsum[r] = 0.0f; }

    const int key_t = tid >> 1;                 // 0..63
    const int hb_t  = (tid & 1) * 32;           // 0 / 32

    for (int kb = 0; kb < wg0 + 64; kb += 64) {
        __syncthreads();
        {
            // ---- stage K tile row-major: global -> LDS direct (async path).
            // IOFFSET is added to BOTH the LDS and global addresses, so one
            // base pair covers the 64B per-thread copy (4 x b128).
            const uint32_t ldsa = (uint32_t)(uintptr_t)&ldsK[key_t][hb_t];
            const uint64_t ga   =
                (uint64_t)(uintptr_t)(kh + (rbase + kb + key_t) * HEADSZ + hb_t);
            asm volatile(
                "global_load_async_to_lds_b128 %0, %1, off\n\t"
                "global_load_async_to_lds_b128 %0, %1, off offset:16\n\t"
                "global_load_async_to_lds_b128 %0, %1, off offset:32\n\t"
                "global_load_async_to_lds_b128 %0, %1, off offset:48"
                :: "v"(ldsa), "v"(ga) : "memory");

            // ---- stage V transposed: ldsVt[head][key] (VGPR path) ----
            const _Float16* vs = vh + (rbase + kb + key_t) * HEADSZ + hb_t;
            v8h vv[4];
#pragma unroll
            for (int j = 0; j < 4; ++j) vv[j] = *(const v8h*)(vs + j * 8);
#pragma unroll
            for (int j = 0; j < 4; ++j)
#pragma unroll
                for (int e = 0; e < 8; ++e)
                    ldsVt[hb_t + j * 8 + e][key_t] = vv[j][e];

            asm volatile("s_wait_asynccnt 0" ::: "memory");
        }
        __syncthreads();

        // ---- S = Q @ K^T : 4 sub-tiles of 16 keys, 2 WMMAs each ----
        v8f s[4];
#pragma unroll
        for (int sub = 0; sub < 4; ++sub) {
            const _Float16* bp = &ldsK[sub * 16 + l15][lhi * 8];
            v16h bk0 = pack16(*(const v8h*)(bp + 0),  *(const v8h*)(bp + 16));
            v16h bk1 = pack16(*(const v8h*)(bp + 32), *(const v8h*)(bp + 48));
            v8f z = (v8f)0.0f;
            z = __builtin_amdgcn_wmma_f32_16x16x32_f16(false, qa0, false, bk0, (short)0, z, false, false);
            z = __builtin_amdgcn_wmma_f32_16x16x32_f16(false, qa1, false, bk1, (short)0, z, false, false);
            s[sub] = z;
        }

        // ---- causal mask ----
#pragma unroll
        for (int sub = 0; sub < 4; ++sub) {
            const int key = kb + sub * 16 + l15;
#pragma unroll
            for (int r = 0; r < 8; ++r)
                if (key > q0 + r + 8 * lhi) s[sub][r] = -1e30f;
        }

        // ---- online softmax: row max (per 16-lane half) ----
        float rmax[8];
#pragma unroll
        for (int r = 0; r < 8; ++r) {
            float v = fmaxf(fmaxf(s[0][r], s[1][r]), fmaxf(s[2][r], s[3][r]));
#pragma unroll
            for (int off = 1; off < 16; off <<= 1)
                v = fmaxf(v, __shfl_xor(v, off, 32));
            rmax[r] = v;
        }
        float mnew[8], alpha[8], rsum[8];
#pragma unroll
        for (int r = 0; r < 8; ++r) {
            mnew[r]  = fmaxf(mrow[r], rmax[r]);
            alpha[r] = __expf(mrow[r] - mnew[r]);
            rsum[r]  = 0.0f;
        }

        // ---- P = exp(S - mnew); stash to LDS (C-layout -> A-layout transpose) ----
#pragma unroll
        for (int sub = 0; sub < 4; ++sub) {
#pragma unroll
            for (int r = 0; r < 8; ++r) {
                float p = __expf(s[sub][r] - mnew[r]);
                rsum[r] += p;
                ldsP[wave][r + 8 * lhi][sub * 16 + l15] = (_Float16)p;
            }
        }
#pragma unroll
        for (int r = 0; r < 8; ++r) {
#pragma unroll
            for (int off = 1; off < 16; off <<= 1)
                rsum[r] += __shfl_xor(rsum[r], off, 32);
            lsum[r] = lsum[r] * alpha[r] + rsum[r];
            mrow[r] = mnew[r];
        }
#pragma unroll
        for (int t = 0; t < 4; ++t)
#pragma unroll
            for (int r = 0; r < 8; ++r) oacc[t][r] *= alpha[r];

        asm volatile("s_wait_dscnt 0" ::: "memory");  // wave-local LDS RAW on ldsP

        const _Float16* pp = &ldsP[wave][l15][lhi * 8];
        v16h pa0 = pack16(*(const v8h*)(pp + 0),  *(const v8h*)(pp + 16));
        v16h pa1 = pack16(*(const v8h*)(pp + 32), *(const v8h*)(pp + 48));

        // ---- O += P @ V : 4 head-tiles, 2 WMMAs each ----
#pragma unroll
        for (int t = 0; t < 4; ++t) {
            const _Float16* vp = &ldsVt[t * 16 + l15][lhi * 8];
            v16h bv0 = pack16(*(const v8h*)(vp + 0),  *(const v8h*)(vp + 16));
            v16h bv1 = pack16(*(const v8h*)(vp + 32), *(const v8h*)(vp + 48));
            oacc[t] = __builtin_amdgcn_wmma_f32_16x16x32_f16(false, pa0, false, bv0, (short)0, oacc[t], false, false);
            oacc[t] = __builtin_amdgcn_wmma_f32_16x16x32_f16(false, pa1, false, bv1, (short)0, oacc[t], false, false);
        }
    }

    // ---- epilogue: normalize and store fp32 ----
#pragma unroll
    for (int t = 0; t < 4; ++t) {
        const int n = t * 16 + l15;
#pragma unroll
        for (int r = 0; r < 8; ++r) {
            const int row = q0 + r + 8 * lhi;
            out[(rbase + row) * HEADSZ + n] = oacc[t][r] / lsum[r];
        }
    }
}

// ---------------------------------------------------------------------------
extern "C" void kernel_launch(void* const* d_in, const int* in_sizes, int n_in,
                              void* d_out, int out_size, void* d_ws, size_t ws_size,
                              hipStream_t stream) {
    (void)in_sizes; (void)n_in; (void)out_size; (void)ws_size;
    const float* x  = (const float*)d_in[0];
    const float* Wq = (const float*)d_in[1];
    const float* bq = (const float*)d_in[2];
    const float* Wk = (const float*)d_in[3];
    const float* bk = (const float*)d_in[4];
    const float* Wv = (const float*)d_in[5];
    const float* bv = (const float*)d_in[6];
    float* out = (float*)d_out;

    _Float16* qh = (_Float16*)d_ws;                 // 8192*64 halves = 1 MB
    _Float16* kh = qh + (size_t)NROWS * HEADSZ;     // +1 MB
    _Float16* vh = kh + (size_t)NROWS * HEADSZ;     // +1 MB

    dim3 blk(128);
    qkv_proj_wmma<<<dim3(NROWS / 64), blk, 0, stream>>>(
        x, Wq, bq, Wk, bk, Wv, bv, qh, kh, vh);
    flash_attn_wmma<<<dim3(NB * (SEQT / 64)), blk, 0, stream>>>(
        qh, kh, vh, out);
}
